// BMLP_773094113632
// MI455X (gfx1250) — matrix operand order repaired and checked by reading, not compiled, not run on hardware
//
#include <hip/hip_runtime.h>

typedef _Float16 h8   __attribute__((ext_vector_type(8)));
typedef _Float16 v16h __attribute__((ext_vector_type(16)));
typedef float    v8f  __attribute__((ext_vector_type(8)));
typedef float    f4   __attribute__((ext_vector_type(4)));
typedef unsigned int u32x4 __attribute__((ext_vector_type(4)));
typedef int          i32x8 __attribute__((ext_vector_type(8)));

#define E_NUM   64
#define NPTS    4096
#define INDIM   16
#define HID     256
#define ROWS_PER_WAVE 16
#define WAVES   8
#define ROWS_PER_BLOCK (ROWS_PER_WAVE * WAVES)   // 128
#define ACT_S   264     // activation row stride in halves (528B -> conflict-free b128)
#define WST     40      // staged weight row stride in halves (80B -> conflict-free b128)

// ---------------- prep: transpose + f32->f16 weights ----------------
// w: [E][K][HID] f32 row-major  ->  wT: [E][HID][Kout] f16, zero-padded K -> Kout
__global__ void prep_transpose(const float* __restrict__ w, _Float16* __restrict__ wT,
                               int K, int Kout, int total) {
    int idx = blockIdx.x * blockDim.x + threadIdx.x;
    if (idx >= total) return;
    int k = idx % Kout;
    int n = (idx / Kout) % HID;
    int e = idx / (Kout * HID);
    float v = (k < K) ? w[((size_t)e * K + k) * HID + n] : 0.0f;
    wT[idx] = (_Float16)v;
}

// ---------------- TDM: DMA one [256 x 32] f16 weight slice -> LDS ----------------
// LDS layout gets +16B padding per 64B row (pad_interval code 3 = 16 DWORDs,
// pad_amount code 3 = 4 DWORDs) => row stride 80B = WST halves.
__device__ __forceinline__ void tdm_issue_wchunk(const _Float16* gsrc, unsigned ldsByteOff, int K) {
    unsigned long long ga = (unsigned long long)(uintptr_t)gsrc;
    u32x4 g0;
    g0[0] = 1u;                                               // count=1, no gather
    g0[1] = ldsByteOff;                                       // lds_addr
    g0[2] = (unsigned)(ga & 0xFFFFFFFFu);                     // global_addr[31:0]
    g0[3] = (unsigned)((ga >> 32) & 0x01FFFFFFu) | (2u << 30); // global_addr[56:32] | type=2
    i32x8 g1;
    g1[0] = (1 << 16)    // data_size = 2 bytes
          | (1 << 20)    // pad_enable
          | (3 << 22)    // pad_interval: 16 DWORDs (64B) between pads
          | (3 << 25);   // pad_amount: 4 DWORDs (16B)
    g1[1] = (K & 0xFFFF) << 16;          // tensor_dim0[15:0] (row length K elems)
    g1[2] = (HID << 16);                 // tensor_dim0[31:16]=0 | tensor_dim1[15:0]=256
    g1[3] = (32 << 16);                  // tensor_dim1[31:16]=0 | tile_dim0=32
    g1[4] = HID;                         // tile_dim1=256, tile_dim2=0
    g1[5] = K;                           // tensor_dim0_stride[31:0]
    g1[6] = 0;                           // tensor_dim0_stride[47:32]=0, dim1_stride lo=0
    g1[7] = 0;
    asm volatile("tensor_load_to_lds %0, %1" :: "s"(g0), "s"(g1) : "memory");
}

// ---------------- one dense layer: [16,K] x [K,256] + bias, SiLU ----------------
__device__ __forceinline__ void dense_layer(
    const _Float16* __restrict__ stageBase,  // LDS, 2 * HID*WST halves (double buffer)
    unsigned stageByteOff,                   // LDS byte offset of stageBase
    const _Float16* __restrict__ wT,         // global, [HID][K] f16 (transposed)
    int K,                                   // 32 or 256
    const float* __restrict__ bias,          // global, [HID] f32
    const _Float16* __restrict__ actIn,      // LDS, this wave's [16][ACT_S]
    _Float16* __restrict__ actOut,           // LDS, this wave's [16][ACT_S]
    int lane, int tid)
{
    const int  nKc    = K >> 5;
    const int  m      = lane & 15;
    const int  aoff   = (lane >> 4) * 8;     // A: K sub-offset per lane half
    const int  koffB  = (lane >> 4) * 8;     // B: K sub-offset per lane half
    const bool wave0  = (tid >> 5) == 0;

    v8f C[16];
    #pragma unroll
    for (int nt = 0; nt < 16; ++nt) {
        v8f z = {0.f, 0.f, 0.f, 0.f, 0.f, 0.f, 0.f, 0.f};
        C[nt] = z;
    }

    if (wave0)
        tdm_issue_wchunk(wT, stageByteOff, K);        // prefetch chunk 0 -> buf 0

    for (int kc = 0; kc < nKc; ++kc) {
        if (wave0) {
            if (kc + 1 < nKc) {
                // kick next chunk into the other buffer, then wait for current
                tdm_issue_wchunk(wT + (size_t)(kc + 1) * 32,
                                 stageByteOff + (unsigned)(((kc + 1) & 1) * (HID * WST * 2)), K);
                __builtin_amdgcn_s_wait_tensorcnt(1);
            } else {
                __builtin_amdgcn_s_wait_tensorcnt(0);
            }
        }
        __syncthreads();

        const _Float16* stage = stageBase + (kc & 1) * (HID * WST);

        // A fragment (ISA 16-bit 16x32 layout)
        const _Float16* arow = actIn + m * ACT_S + kc * 32;
        h8 alo = *(const h8*)(arow + aoff);
        h8 ahi = *(const h8*)(arow + 16 + aoff);
        v16h A;
        #pragma unroll
        for (int i = 0; i < 8; ++i) { A[i] = alo[i]; A[i + 8] = ahi[i]; }

        #pragma unroll
        for (int nt = 0; nt < 16; ++nt) {
            const _Float16* brow = stage + (nt * 16 + m) * WST + koffB;
            h8 blo = *(const h8*)brow;
            h8 bhi = *(const h8*)(brow + 16);
            v16h B;
            #pragma unroll
            for (int i = 0; i < 8; ++i) { B[i] = blo[i]; B[i + 8] = bhi[i]; }
            C[nt] = __builtin_amdgcn_wmma_f32_16x16x32_f16(
                false, A, false, B, (short)0, C[nt], false, false);
        }
        __syncthreads();   // all reads of this buffer done before TDM may overwrite it
    }

    // epilogue: bias + SiLU, write f16 activations back to LDS
    const int rowoff = (lane >> 4) * 8;
    #pragma unroll
    for (int nt = 0; nt < 16; ++nt) {
        const int n = nt * 16 + m;
        const float bn = bias[n];
        #pragma unroll
        for (int i = 0; i < 8; ++i) {
            float v = C[nt][i] + bn;
            float s = __builtin_amdgcn_rcpf(1.0f + __expf(-v));
            actOut[(rowoff + i) * ACT_S + n] = (_Float16)(v * s);
        }
    }
}

// ---------------- main kernel ----------------
__global__ __launch_bounds__(256)
void bmlp_kernel(const float* __restrict__ x,
                 const float* __restrict__ b0,
                 const float* __restrict__ b1,
                 const float* __restrict__ b2,
                 const float* __restrict__ wl,
                 const float* __restrict__ bl,
                 const _Float16* __restrict__ w0T,
                 const _Float16* __restrict__ w1T,
                 const _Float16* __restrict__ w2T,
                 float* __restrict__ out)
{
    __shared__ _Float16 stage[2 * HID * WST];                      // 40 KB (double buffer)
    __shared__ _Float16 act[2 * WAVES * ROWS_PER_WAVE * ACT_S];    // 132 KB

    const int tid  = threadIdx.x;
    const int lane = tid & 31;
    const int wave = tid >> 5;
    const int tile = blockIdx.x & 31;        // NPTS / 128 = 32 tiles
    const int e    = blockIdx.x >> 5;
    const int p0   = tile * ROWS_PER_BLOCK + wave * ROWS_PER_WAVE;

    const unsigned stageOff = (unsigned)(uintptr_t)(&stage[0]);   // LDS byte offset

    _Float16* buf0 = act + wave * (ROWS_PER_WAVE * ACT_S);
    _Float16* buf1 = act + WAVES * ROWS_PER_WAVE * ACT_S + wave * (ROWS_PER_WAVE * ACT_S);

    // load x tile [16,16] f32 -> buf0 as f16, zero-pad K 16..31
    {
        const int m   = lane >> 1;
        const int col = (lane & 1) * 8;
        const float* xp = x + ((size_t)e * NPTS + p0 + m) * INDIM + col;
        f4 x0 = *(const f4*)xp;
        f4 x1 = *(const f4*)(xp + 4);
        h8 hx;
        #pragma unroll
        for (int i = 0; i < 4; ++i) { hx[i] = (_Float16)x0[i]; hx[4 + i] = (_Float16)x1[i]; }
        *(h8*)(buf0 + m * ACT_S + col) = hx;
        h8 z = {};
        *(h8*)(buf0 + m * ACT_S + 16 + col) = z;
    }

    dense_layer(stage, stageOff, w0T + (size_t)e * HID * 32,  32,  b0 + e * HID, buf0, buf1, lane, tid);
    dense_layer(stage, stageOff, w1T + (size_t)e * HID * HID, 256, b1 + e * HID, buf1, buf0, lane, tid);
    dense_layer(stage, stageOff, w2T + (size_t)e * HID * HID, 256, b2 + e * HID, buf0, buf1, lane, tid);

    // final layer: [16,256] . [256,1] + bl
    {
        const int m    = lane & 15;
        const int half = lane >> 4;
        const _Float16* a = buf1 + m * ACT_S + half * 128;
        const float* wp   = wl + (size_t)e * HID + half * 128;
        float sum = 0.f;
        #pragma unroll
        for (int k = 0; k < 128; k += 8) {
            h8 av = *(const h8*)(a + k);
            f4 w0v = *(const f4*)(wp + k);
            f4 w1v = *(const f4*)(wp + k + 4);
            sum += (float)av[0] * w0v[0] + (float)av[1] * w0v[1]
                 + (float)av[2] * w0v[2] + (float)av[3] * w0v[3]
                 + (float)av[4] * w1v[0] + (float)av[5] * w1v[1]
                 + (float)av[6] * w1v[2] + (float)av[7] * w1v[3];
        }
        float other = __shfl_down(sum, 16, 32);
        if (half == 0)
            out[(size_t)e * NPTS + p0 + m] = sum + other + bl[e];
    }
}

extern "C" void kernel_launch(void* const* d_in, const int* in_sizes, int n_in,
                              void* d_out, int out_size, void* d_ws, size_t ws_size,
                              hipStream_t stream) {
    const float* x  = (const float*)d_in[0];
    const float* w0 = (const float*)d_in[1];
    const float* b0 = (const float*)d_in[2];
    const float* w1 = (const float*)d_in[3];
    const float* b1 = (const float*)d_in[4];
    const float* w2 = (const float*)d_in[5];
    const float* b2 = (const float*)d_in[6];
    const float* wl = (const float*)d_in[7];
    const float* bl = (const float*)d_in[8];
    float* out = (float*)d_out;

    _Float16* w0T = (_Float16*)d_ws;                       // E*256*32 halves (1 MB)
    _Float16* w1T = w0T + (size_t)E_NUM * HID * 32;        // E*256*256 halves (8 MB)
    _Float16* w2T = w1T + (size_t)E_NUM * HID * HID;       // E*256*256 halves (8 MB)

    const int t0 = E_NUM * HID * 32;
    const int t1 = E_NUM * HID * HID;
    prep_transpose<<<(t0 + 255) / 256, 256, 0, stream>>>(w0, w0T, INDIM, 32, t0);
    prep_transpose<<<(t1 + 255) / 256, 256, 0, stream>>>(w1, w1T, HID, HID, t1);
    prep_transpose<<<(t1 + 255) / 256, 256, 0, stream>>>(w2, w2T, HID, HID, t1);

    bmlp_kernel<<<E_NUM * (NPTS / ROWS_PER_BLOCK), 256, 0, stream>>>(
        x, b0, b1, b2, wl, bl, w0T, w1T, w2T, out);
}
